// Decoder_70909910056982
// MI455X (gfx1250) — matrix operand order repaired and checked by reading, not compiled, not run on hardware
//
#include <hip/hip_runtime.h>
#include <hip/hip_bf16.h>
#include <math.h>

// Problem constants (from the reference): L=4, B=1, S=2048, E=768, N=12, H=64, FF=3072
#define LAYERS 4
#define SEQ    2048
#define EMB    768
#define NHEAD  12
#define HDIM   64
#define FFDIM  3072

typedef __attribute__((ext_vector_type(16))) _Float16 v16h;
typedef __attribute__((ext_vector_type(8)))  _Float16 v8h;
typedef __attribute__((ext_vector_type(8)))  float    v8f;
typedef __attribute__((ext_vector_type(4)))  unsigned int u32x4;
typedef __attribute__((ext_vector_type(4)))  int      i32x4;
typedef __attribute__((ext_vector_type(8)))  int      i32x8;

union ABfrag { v16h v; v8h h[2]; };

// --- CDNA5 Tensor Data Mover availability (probe-verified builtin) ----------
#if defined(__AMDGCN__) && __has_builtin(__builtin_amdgcn_tensor_load_to_lds)
#define HAVE_TDM 1
#else
#define HAVE_TDM 0
#endif
#if HAVE_TDM && __has_include(<hip/amd_detail/amd_gfx1250_TDM.h>)
#define TDM_SIX_ARGS 1   // therock/clang-23 toolchain: 6-arg form
#else
#define TDM_SIX_ARGS 0   // ROCm 7.2 / clang-22: 5-arg form
#endif

__device__ __forceinline__ v8f zero8f() {
  v8f z = {0.f, 0.f, 0.f, 0.f, 0.f, 0.f, 0.f, 0.f};
  return z;
}

__device__ __forceinline__ v8f wmma_f16(v16h a, v16h b, v8f c) {
  // D = A(16x32 f16) * B(32x16 f16) + C(16x16 f32)
  return __builtin_amdgcn_wmma_f32_16x16x32_f16(false, a, false, b, (short)0, c,
                                                false, false);
}

// ---------------------------------------------------------------------------
// TDM: DMA a 128x32-half 2D tile (row stride = strideElems halfs) from global
// into LDS, with LDS padding 4 DWORDs every 16 DWORDs so rows land at the
// conflict-free LDA=40-half stride.  D# bitfields per CDNA5 ISA ch.8.
// ---------------------------------------------------------------------------
#if HAVE_TDM
__device__ __forceinline__ void tdm_load_a_tile(const _Float16* gsrc,
                                                unsigned int lds_byte_off,
                                                int strideElems) {
  unsigned long long ga = (unsigned long long)(uintptr_t)gsrc;
  u32x4 g0;
  g0[0] = 1u;                                   // count=1 (valid user D#)
  g0[1] = lds_byte_off;                         // lds_addr (bytes)
  g0[2] = (unsigned int)ga;                     // global_addr[31:0]
  g0[3] = (unsigned int)((ga >> 32) & 0x1FFFFFFull) | (2u << 30); // [56:32]|type=2
  i32x8 g1;
  g1[0] = (1 << 16)        // data_size = 2 bytes
        | (1 << 20)        // pad_enable
        | (3 << 22)        // pad_interval: 16 DWORDs (one 32-half row)
        | (3 << 25);       // pad_amount: 4 DWORDs (8 halfs) -> LDA = 40
  g1[1] = (strideElems & 0xFFFF) << 16;                          // tensor_dim0 lo
  g1[2] = ((strideElems >> 16) & 0xFFFF) | ((SEQ & 0xFFFF) << 16); // dim0 hi|dim1 lo
  g1[3] = (32 << 16);                                            // tile_dim0 = 32
  g1[4] = 128;                                                   // tile_dim1 = 128
  g1[5] = strideElems;                                           // dim0_stride lo
  g1[6] = 0;
  g1[7] = 0;
  i32x4 z4; z4[0] = z4[1] = z4[2] = z4[3] = 0;
#if TDM_SIX_ARGS
  i32x8 z8;
  z8[0]=z8[1]=z8[2]=z8[3]=z8[4]=z8[5]=z8[6]=z8[7]=0;
  __builtin_amdgcn_tensor_load_to_lds(g0, g1, z4, z4, z8, 0);
#else
  __builtin_amdgcn_tensor_load_to_lds(g0, g1, z4, z4, 0);
#endif
}
#endif

// ---------------------------------------------------------------------------
// fp32 -> fp16 conversion (weights)
// ---------------------------------------------------------------------------
__global__ __launch_bounds__(256) void cvt_kernel(const float* __restrict__ in,
                                                  _Float16* __restrict__ out, int n) {
  int i = blockIdx.x * 256 + threadIdx.x;
  if (i < n) out[i] = (_Float16)in[i];
}

// ---------------------------------------------------------------------------
// V transpose: qkv[s, 2E + e] -> vt[e, s]   (e = head*H + h)
// ---------------------------------------------------------------------------
__global__ __launch_bounds__(256) void vtrans_kernel(const _Float16* __restrict__ qkv,
                                                     _Float16* __restrict__ vt) {
  int i = blockIdx.x * 256 + threadIdx.x;   // over EMB*SEQ, s fastest
  if (i >= EMB * SEQ) return;
  int e = i / SEQ, s = i - e * SEQ;
  vt[i] = qkv[(size_t)s * (3 * EMB) + 2 * EMB + e];
}

// ---------------------------------------------------------------------------
// LayerNorm over E=768 per row; writes fp32 residual stream + fp16 GEMM input
// ---------------------------------------------------------------------------
__global__ __launch_bounds__(256) void ln_kernel(const float* __restrict__ in,
                                                 const float* __restrict__ w,
                                                 const float* __restrict__ b,
                                                 float* __restrict__ out,
                                                 _Float16* __restrict__ out16) {
  __shared__ float red[256];
  __shared__ float stats[2];
  const int row = blockIdx.x, tid = threadIdx.x;
  const float* x = in + (size_t)row * EMB;

  float s = 0.f;
  for (int i = tid; i < EMB; i += 256) s += x[i];
  red[tid] = s;
  __syncthreads();
  for (int off = 128; off; off >>= 1) {
    if (tid < off) red[tid] += red[tid + off];
    __syncthreads();
  }
  if (tid == 0) stats[0] = red[0] * (1.f / EMB);
  __syncthreads();
  const float mean = stats[0];

  float v = 0.f;
  for (int i = tid; i < EMB; i += 256) {
    float d = x[i] - mean;
    v += d * d;
  }
  red[tid] = v;
  __syncthreads();
  for (int off = 128; off; off >>= 1) {
    if (tid < off) red[tid] += red[tid + off];
    __syncthreads();
  }
  if (tid == 0) stats[1] = rsqrtf(red[0] * (1.f / EMB) + 1e-5f);
  __syncthreads();
  const float rstd = stats[1];

  for (int i = tid; i < EMB; i += 256) {
    float o = (x[i] - mean) * rstd * w[i] + b[i];
    out[(size_t)row * EMB + i] = o;
    out16[(size_t)row * EMB + i] = (_Float16)o;
  }
}

// ---------------------------------------------------------------------------
// WMMA GEMM: C[M,N] = A[M,K](f16) * W[K,N](f16) + bias, fused epilogue.
// MODE 0: store f16          (QKV projection)
// MODE 1: GELU(exact) -> f16 (MLP fc)
// MODE 2: + residual -> f32  (attn proj / MLP proj, pre-LN sum)
// Workgroup tile 128x128, 8 waves (4x2), wave tile 32x64 (2x4 WMMA tiles).
// Double-buffered LDS; A tile staged by the Tensor Data Mover (TDM) with
// hardware LDS padding; B tile transposed via ds scatter.
// ---------------------------------------------------------------------------
#define BM 128
#define BN 128
#define BK 32
#define LDA 40   // padded LDS stride (halfs): conflict-free frag reads
#define LDB 40

template <int MODE>
__global__ __launch_bounds__(256) void gemm_wmma_kernel(
    const _Float16* __restrict__ A, const _Float16* __restrict__ W,
    const float* __restrict__ bias, const float* __restrict__ residual,
    _Float16* __restrict__ out16, float* __restrict__ out32,
    const int K, const int N) {
  __shared__ __align__(16) _Float16 As[2][BM * LDA];
  __shared__ __align__(16) _Float16 Bs[2][BN * LDB];

  const int tid = threadIdx.x;
  const int lane = tid & 31, wave = tid >> 5;
  const int wm = wave & 3, wn = wave >> 2;   // 4x2 wave grid
  const int ll = lane & 15, lh = lane >> 4;  // lane halves (WMMA striping)
  const int blockN = blockIdx.x * BN;
  const int blockM = blockIdx.y * BM;
  const int NT = K / BK;

  v8f acc[2][4];
#pragma unroll
  for (int i = 0; i < 2; ++i)
#pragma unroll
    for (int j = 0; j < 4; ++j) acc[i][j] = zero8f();

  auto stageA = [&](int t, int buf) {
#if HAVE_TDM
    if (wave == 0)
      tdm_load_a_tile(&A[(size_t)blockM * K + t * BK],
                      (unsigned int)(uintptr_t)&As[buf][0], K);
#else
#pragma unroll
    for (int i = 0; i < 2; ++i) {
      int c = tid + i * 256;        // 512 chunks of 8 halfs
      int row = c >> 2;
      int col = (c & 3) * 8;
      *(v8h*)&As[buf][row * LDA + col] =
          *(const v8h*)&A[(size_t)(blockM + row) * K + t * BK + col];
    }
#endif
  };
  auto stageB = [&](int t, int buf) {
#pragma unroll
    for (int i = 0; i < 2; ++i) {
      int c = tid + i * 256;        // 32 k-rows x 16 chunks-of-8
      int kk = c >> 4;
      int nn = (c & 15) * 8;
      v8h d = *(const v8h*)&W[(size_t)(t * BK + kk) * N + blockN + nn];
#pragma unroll
      for (int e = 0; e < 8; ++e) Bs[buf][(nn + e) * LDB + kk] = d[e];
    }
  };

  // prologue: tile 0 into buffer 0
  stageA(0, 0);
  stageB(0, 0);
#if HAVE_TDM
  if (wave == 0) __builtin_amdgcn_s_wait_tensorcnt(0);
#endif
  __syncthreads();

  for (int t = 0; t < NT; ++t) {
    const int cur = t & 1, nxt = cur ^ 1;
    if (t + 1 < NT) {          // overlap next-tile DMA/staging with compute
      stageA(t + 1, nxt);
      stageB(t + 1, nxt);
    }
    if (t + 2 < NT)            // warm L2 for the tile after next
      __builtin_prefetch(&W[(size_t)((t + 2) * BK + (tid >> 3)) * N + blockN], 0, 0);

    // ---- fragments: A layout (lane=row m, K split per lane-half), B layout
    //      (lane=col n, same K split as A) -> all ds_load_b128 ----
    ABfrag af[2], bf[4];
#pragma unroll
    for (int i = 0; i < 2; ++i) {
      int row = wm * 32 + i * 16 + ll;
      af[i].h[0] = *(const v8h*)&As[cur][row * LDA + lh * 8];       // K=kb..kb+7
      af[i].h[1] = *(const v8h*)&As[cur][row * LDA + lh * 8 + 16];  // K=kb+16..+23
    }
#pragma unroll
    for (int j = 0; j < 4; ++j) {
      int n = wn * 64 + j * 16 + ll;
      bf[j].h[0] = *(const v8h*)&Bs[cur][n * LDB + lh * 8];
      bf[j].h[1] = *(const v8h*)&Bs[cur][n * LDB + lh * 8 + 16];
    }
#pragma unroll
    for (int i = 0; i < 2; ++i)
#pragma unroll
      for (int j = 0; j < 4; ++j)
        acc[i][j] = wmma_f16(af[i].v, bf[j].v, acc[i][j]);

#if HAVE_TDM
    if (wave == 0) __builtin_amdgcn_s_wait_tensorcnt(0);  // next A tile landed
#endif
    __syncthreads();
  }

  // ---- epilogue: C layout (lane=col n; rows r + lh*8 across 8 VGPRs) ----
#pragma unroll
  for (int i = 0; i < 2; ++i) {
    const int gm0 = blockM + wm * 32 + i * 16 + lh * 8;
#pragma unroll
    for (int j = 0; j < 4; ++j) {
      const int gn = blockN + wn * 64 + j * 16 + ll;
      const float bv = bias[gn];
#pragma unroll
      for (int r = 0; r < 8; ++r) {
        const int gm = gm0 + r;
        float v = acc[i][j][r] + bv;
        if (MODE == 1) v = 0.5f * v * (1.f + erff(v * 0.70710678118f));
        if (MODE == 2) {
          v += residual[(size_t)gm * N + gn];
          out32[(size_t)gm * N + gn] = v;
        } else {
          out16[(size_t)gm * N + gn] = (_Float16)v;
        }
      }
    }
  }
}

// ---------------------------------------------------------------------------
// Flash attention (causal), WMMA for QK^T and PV.
// Grid: (S/64, NHEAD), block 128 = 4 waves; each wave owns 16 query rows.
// Key blocks of 64; online softmax; P transposed to A-layout via per-wave LDS.
// ---------------------------------------------------------------------------
__global__ __launch_bounds__(128) void attn_kernel(const _Float16* __restrict__ qkv,
                                                   const _Float16* __restrict__ vt,
                                                   _Float16* __restrict__ y) {
  const int qblk = blockIdx.x, head = blockIdx.y;
  const int wave = threadIdx.x >> 5, lane = threadIdx.x & 31;
  const int ll = lane & 15, lh = lane >> 4;
  const int qbase = qblk * 64 + wave * 16;
  const int LP = 72;  // padded stride (halfs), 144B = 16B-aligned rows
  __shared__ __align__(16) _Float16 Pl[4][16 * 72];
  _Float16* P = &Pl[wave][0];

  // Q A-frags (2 K-steps over H=64), held in registers for the whole pass
  ABfrag qf[2];
  {
    const _Float16* qp = qkv + (size_t)(qbase + ll) * (3 * EMB) + head * HDIM;
#pragma unroll
    for (int ks = 0; ks < 2; ++ks) {
      qf[ks].h[0] = *(const v8h*)&qp[ks * 32 + lh * 8];
      qf[ks].h[1] = *(const v8h*)&qp[ks * 32 + lh * 8 + 16];
    }
  }

  v8f o[4];
#pragma unroll
  for (int j = 0; j < 4; ++j) o[j] = zero8f();
  float mstat[8], lstat[8];
#pragma unroll
  for (int r = 0; r < 8; ++r) { mstat[r] = -1e30f; lstat[r] = 0.f; }

  for (int kb = 0; kb <= qblk; ++kb) {  // causal: skip key blocks beyond diag
    const int t0 = kb * 64;

    // ---- scores: S(16x64) = Q(16x64) @ K^T, 4 n-tiles x 2 K-steps ----
    v8f s[4];
#pragma unroll
    for (int j = 0; j < 4; ++j) {
      s[j] = zero8f();
      const _Float16* kp = qkv + (size_t)(t0 + j * 16 + ll) * (3 * EMB) + EMB +
                           head * HDIM + lh * 8;
#pragma unroll
      for (int ks = 0; ks < 2; ++ks) {
        ABfrag kf;
        kf.h[0] = *(const v8h*)&kp[ks * 32];
        kf.h[1] = *(const v8h*)&kp[ks * 32 + 16];
        s[j] = wmma_f16(qf[ks].v, kf.v, s[j]);
      }
    }

    // ---- scale + causal mask + running row max ----
    float rmax[8];
#pragma unroll
    for (int r = 0; r < 8; ++r) rmax[r] = -1e30f;
#pragma unroll
    for (int j = 0; j < 4; ++j) {
      const int key = t0 + j * 16 + ll;
#pragma unroll
      for (int r = 0; r < 8; ++r) {
        float sv = s[j][r] * 0.125f;  // 1/sqrt(H)
        if (key > qbase + r + lh * 8) sv = -1e30f;
        s[j][r] = sv;
        rmax[r] = fmaxf(rmax[r], sv);
      }
    }
#pragma unroll
    for (int r = 0; r < 8; ++r) {   // 16-lane row reduction (C-layout halves)
      float m = rmax[r];
      m = fmaxf(m, __shfl_xor(m, 1));
      m = fmaxf(m, __shfl_xor(m, 2));
      m = fmaxf(m, __shfl_xor(m, 4));
      m = fmaxf(m, __shfl_xor(m, 8));
      float mn = fmaxf(mstat[r], m);
      float alpha = __expf(mstat[r] - mn);
      mstat[r] = mn;
      lstat[r] *= alpha;
#pragma unroll
      for (int j = 0; j < 4; ++j) o[j][r] *= alpha;
    }

    // ---- P = exp(S - m), row sums ----
    float rsum[8];
#pragma unroll
    for (int r = 0; r < 8; ++r) rsum[r] = 0.f;
#pragma unroll
    for (int j = 0; j < 4; ++j)
#pragma unroll
      for (int r = 0; r < 8; ++r) {
        float p = __expf(s[j][r] - mstat[r]);
        s[j][r] = p;
        rsum[r] += p;
      }
#pragma unroll
    for (int r = 0; r < 8; ++r) {
      float t = rsum[r];
      t += __shfl_xor(t, 1);
      t += __shfl_xor(t, 2);
      t += __shfl_xor(t, 4);
      t += __shfl_xor(t, 8);
      lstat[r] += t;
    }

    // ---- transpose P (C-layout) -> A-layout via per-wave LDS ----
#pragma unroll
    for (int j = 0; j < 4; ++j)
#pragma unroll
      for (int r = 0; r < 8; ++r)
        P[(r + lh * 8) * LP + j * 16 + ll] = (_Float16)s[j][r];
    __builtin_amdgcn_wave_barrier();
    asm volatile("s_wait_dscnt 0x0" ::: "memory");  // in-wave LDS visibility

    // ---- O += P(16x64) @ V(64x64): 2 K-steps x 4 h-tiles ----
#pragma unroll
    for (int ks = 0; ks < 2; ++ks) {
      ABfrag pf;
      pf.h[0] = *(const v8h*)&P[ll * LP + ks * 32 + lh * 8];
      pf.h[1] = *(const v8h*)&P[ll * LP + ks * 32 + lh * 8 + 16];
#pragma unroll
      for (int j = 0; j < 4; ++j) {
        ABfrag vf;
        const _Float16* vp = vt + (size_t)(head * HDIM + j * 16 + ll) * SEQ + t0 +
                             ks * 32 + lh * 8;
        vf.h[0] = *(const v8h*)&vp[0];
        vf.h[1] = *(const v8h*)&vp[16];
        o[j] = wmma_f16(pf.v, vf.v, o[j]);
      }
    }
    __builtin_amdgcn_wave_barrier();
  }

  // ---- normalize and store fp16 attention output ----
#pragma unroll
  for (int j = 0; j < 4; ++j)
#pragma unroll
    for (int r = 0; r < 8; ++r) {
      float val = o[j][r] / lstat[r];
      int row = qbase + r + lh * 8;
      y[(size_t)row * EMB + head * HDIM + j * 16 + ll] = (_Float16)val;
    }
}

// ---------------------------------------------------------------------------
// Host-side orchestration
// ---------------------------------------------------------------------------
extern "C" void kernel_launch(void* const* d_in, const int* in_sizes, int n_in,
                              void* d_out, int out_size, void* d_ws, size_t ws_size,
                              hipStream_t stream) {
  (void)in_sizes; (void)n_in; (void)out_size; (void)ws_size;
  const float* x        = (const float*)d_in[0];
  // d_in[1] = mask (causal; encoded directly in the attention kernel)
  const float* c_attn_w = (const float*)d_in[2];
  const float* c_attn_b = (const float*)d_in[3];
  const float* c_proj_w = (const float*)d_in[4];
  const float* c_proj_b = (const float*)d_in[5];
  const float* ln1_w    = (const float*)d_in[6];
  const float* ln1_b    = (const float*)d_in[7];
  const float* fc_w     = (const float*)d_in[8];
  const float* fc_b     = (const float*)d_in[9];
  const float* proj_w   = (const float*)d_in[10];
  const float* proj_b   = (const float*)d_in[11];
  const float* ln2_w    = (const float*)d_in[12];
  const float* ln2_b    = (const float*)d_in[13];
  const float* lnf_w    = (const float*)d_in[14];
  const float* lnf_b    = (const float*)d_in[15];
  float* out = (float*)d_out;

  char* p = (char*)d_ws;
  auto carve = [&p](size_t bytes) {
    char* r = p;
    p += (bytes + 255) & ~(size_t)255;
    return r;
  };
  _Float16* wqkv16 = (_Float16*)carve((size_t)LAYERS * EMB * 3 * EMB * 2);
  _Float16* wprj16 = (_Float16*)carve((size_t)LAYERS * EMB * EMB * 2);
  _Float16* wfc16  = (_Float16*)carve((size_t)LAYERS * EMB * FFDIM * 2);
  _Float16* wo16   = (_Float16*)carve((size_t)LAYERS * FFDIM * EMB * 2);
  float*    h      = (float*)carve((size_t)SEQ * EMB * 4);
  float*    res    = (float*)carve((size_t)SEQ * EMB * 4);
  _Float16* h16    = (_Float16*)carve((size_t)SEQ * EMB * 2);
  _Float16* qkv16  = (_Float16*)carve((size_t)SEQ * 3 * EMB * 2);
  _Float16* vt16   = (_Float16*)carve((size_t)EMB * SEQ * 2);
  _Float16* y16    = (_Float16*)carve((size_t)SEQ * EMB * 2);
  _Float16* act16  = (_Float16*)carve((size_t)SEQ * FFDIM * 2);

  auto cvt = [&](const float* src, _Float16* dst, size_t n) {
    cvt_kernel<<<dim3((unsigned)((n + 255) / 256)), dim3(256), 0, stream>>>(
        src, dst, (int)n);
  };
  cvt(c_attn_w, wqkv16, (size_t)LAYERS * EMB * 3 * EMB);
  cvt(c_proj_w, wprj16, (size_t)LAYERS * EMB * EMB);
  cvt(fc_w,     wfc16,  (size_t)LAYERS * EMB * FFDIM);
  cvt(proj_w,   wo16,   (size_t)LAYERS * FFDIM * EMB);

  // Final LayerNorm is applied FIRST (reference composes right-to-left)
  ln_kernel<<<dim3(SEQ), dim3(256), 0, stream>>>(x, lnf_w, lnf_b, h, h16);

  // scan(reverse=True): blocks applied from layer L-1 down to 0
  for (int l = LAYERS - 1; l >= 0; --l) {
    gemm_wmma_kernel<0><<<dim3(3 * EMB / BN, SEQ / BM), dim3(256), 0, stream>>>(
        h16, wqkv16 + (size_t)l * EMB * 3 * EMB, c_attn_b + (size_t)l * 3 * EMB,
        nullptr, qkv16, nullptr, EMB, 3 * EMB);
    vtrans_kernel<<<dim3((EMB * SEQ) / 256), dim3(256), 0, stream>>>(qkv16, vt16);
    attn_kernel<<<dim3(SEQ / 64, NHEAD), dim3(128), 0, stream>>>(qkv16, vt16, y16);
    gemm_wmma_kernel<2><<<dim3(EMB / BN, SEQ / BM), dim3(256), 0, stream>>>(
        y16, wprj16 + (size_t)l * EMB * EMB, c_proj_b + (size_t)l * EMB,
        h, nullptr, res, EMB, EMB);
    ln_kernel<<<dim3(SEQ), dim3(256), 0, stream>>>(
        res, ln1_w + (size_t)l * EMB, ln1_b + (size_t)l * EMB, h, h16);
    gemm_wmma_kernel<1><<<dim3(FFDIM / BN, SEQ / BM), dim3(256), 0, stream>>>(
        h16, wfc16 + (size_t)l * EMB * FFDIM, fc_b + (size_t)l * FFDIM,
        nullptr, act16, nullptr, EMB, FFDIM);
    gemm_wmma_kernel<2><<<dim3(EMB / BN, SEQ / BM), dim3(256), 0, stream>>>(
        act16, wo16 + (size_t)l * FFDIM * EMB, proj_b + (size_t)l * EMB,
        h, nullptr, res, FFDIM, EMB);
    float* lnout = (l == 0) ? out : h;  // last LN writes the kernel output
    ln_kernel<<<dim3(SEQ), dim3(256), 0, stream>>>(
        res, ln2_w + (size_t)l * EMB, ln2_b + (size_t)l * EMB, lnout, h16);
  }
}
